// Model_76347338654046
// MI455X (gfx1250) — matrix operand order repaired and checked by reading, not compiled, not run on hardware
//
#include <hip/hip_runtime.h>
#include <math.h>

// ---------------------------------------------------------------------------
// Problem constants (from reference): N=8192, H=512, C=100, R=69, P=69
// ---------------------------------------------------------------------------
#define NROWS 8192
#define HDIM  512
#define CCAT  100
#define RREL  69
#define PDIM  69
#define EPSV  1e-8f

typedef __bf16 bhalf;
typedef bhalf v16bf __attribute__((ext_vector_type(16)));
typedef float v8f   __attribute__((ext_vector_type(8)));
typedef unsigned int u32;
typedef u32 u32x4 __attribute__((ext_vector_type(4)));

__device__ __forceinline__ bhalf f2bf(float x) { return (bhalf)x; }

// ---------------------------------------------------------------------------
// Weight prep: f32 (K x Nout, row-major) -> bf16 TRANSPOSED (Npad x K),
// zero-padded rows for n in [Nout, Npad).
// ---------------------------------------------------------------------------
__global__ void k_transpose_bf16(const float* __restrict__ in,
                                 bhalf* __restrict__ out,
                                 int K, int Nout, int Npad) {
  long total = (long)Npad * K;
  long i = (long)blockIdx.x * blockDim.x + threadIdx.x;
  long stride = (long)gridDim.x * blockDim.x;
  for (; i < total; i += stride) {
    int n = (int)(i / K);
    int k = (int)(i - (long)n * K);
    float v = (n < Nout) ? in[(size_t)k * Nout + n] : 0.0f;
    out[i] = f2bf(v);
  }
}

// ---------------------------------------------------------------------------
// concat([s,p,o], axis=1) with f32 -> bf16 convert: out is (N, 1536) bf16
// ---------------------------------------------------------------------------
__global__ void k_concat_bf16(const float* __restrict__ s,
                              const float* __restrict__ p,
                              const float* __restrict__ o,
                              bhalf* __restrict__ out) {
  const long total = (long)NROWS * 1536;
  long i = (long)blockIdx.x * blockDim.x + threadIdx.x;
  long stride = (long)gridDim.x * blockDim.x;
  for (; i < total; i += stride) {
    long n = i / 1536;
    int  j = (int)(i - n * 1536);
    float v = (j < 512) ? s[n * 512 + j]
            : (j < 1024) ? p[n * 512 + (j - 512)]
                         : o[n * 512 + (j - 1024)];
    out[i] = f2bf(v);
  }
}

// ---------------------------------------------------------------------------
// bf16 WMMA GEMM with async double-buffered LDS staging.
//   out = act(A[MxK] @ B[KxNout] + bias), f32 accumulate.
//  - B pre-transposed + padded: Bt[Npad x K], Npad % 64 == 0.
//  - 256 threads = 8 wave32s; block tile 128(M) x 64(N); K-step 32.
//  - Staging: GLOBAL_LOAD_ASYNC_TO_LDS_B128 (ASYNCcnt) into ping-pong LDS
//    buffers; tile k+1 DMA overlaps tile k WMMAs. INST_OFFSET applies to
//    both the LDS and global address, so the 32B A-chunk uses offset:16.
//  - Per K-step: load A frag + all 4 B frags into distinct registers FIRST
//    (one ds_load clause, partial dscnt waits), then 4 back-to-back WMMAs.
// ---------------------------------------------------------------------------
__global__ __launch_bounds__(256) void k_gemm_bf16(
    const bhalf* __restrict__ A, const bhalf* __restrict__ Bt,
    const float* __restrict__ bias,
    float* __restrict__ outF, bhalf* __restrict__ outB,
    int M, int K, int Nout, int relu)
{
  __shared__ bhalf As[2][128][40];  // [buf][m][k], pad 32->40 keeps 16B align
  __shared__ bhalf Bs[2][64][40];   // [buf][n][k] (B already transposed)

  const int tid  = threadIdx.x;
  const int wave = tid >> 5;        // 0..7 : m-tile index
  const int lane = tid & 31;
  const int m0 = blockIdx.y * 128;
  const int n0 = blockIdx.x * 64;

  const int half = (lane < 16) ? 0 : 1;
  const int l16  = lane & 15;

  const int arow  = wave * 16 + l16;   // A fragment row in LDS
  const int abase = half * 16;         // A fragment K base (lanes 0-15 / 16-31)
  const int kb2   = half * 16;         // B fragment K base

  v8f acc[4] = {v8f{}, v8f{}, v8f{}, v8f{}};

  // staging assignments (constant per thread)
  const int sArow = tid >> 1;          // 0..127, 32B each (two b128 chunks)
  const int sAseg = (tid & 1) * 16;    // element offset (16 bf16 = 32B)
  const int sBrow = tid >> 2;          // 0..63,  16B each
  const int sBseg = (tid & 3) * 8;

  const bhalf* gA = A  + (size_t)(m0 + sArow) * K + sAseg;
  const bhalf* gB = Bt + (size_t)(n0 + sBrow) * K + sBseg;

  // issue async staging for tile at k0 into buffer `buf`
  auto issue_stage = [&](int buf, int k0) {
    unsigned la = (unsigned)(uintptr_t)&As[buf][sArow][sAseg];
    unsigned lb = (unsigned)(uintptr_t)&Bs[buf][sBrow][sBseg];
    unsigned long long ga = (unsigned long long)(uintptr_t)(gA + k0);
    unsigned long long gb = (unsigned long long)(uintptr_t)(gB + k0);
    asm volatile("global_load_async_to_lds_b128 %0, %1, off"
                 :: "v"(la), "v"(ga) : "memory");
    asm volatile("global_load_async_to_lds_b128 %0, %1, off offset:16"
                 :: "v"(la), "v"(ga) : "memory");
    asm volatile("global_load_async_to_lds_b128 %0, %1, off"
                 :: "v"(lb), "v"(gb) : "memory");
  };

  issue_stage(0, 0);

  for (int k0 = 0; k0 < K; k0 += 32) {
    const int cur = (k0 >> 5) & 1;

    // wait for this tile's async DMA, make it visible to all waves
    asm volatile("s_wait_asynccnt 0x0" ::: "memory");
    __syncthreads();

    // overlap next tile's DMA with this tile's WMMAs (other buffer's readers
    // all passed the previous iteration's end barrier)
    if (k0 + 32 < K) issue_stage(cur ^ 1, k0 + 32);

    // ---- load A fragment + all 4 B fragments into distinct registers ----
    union F { v16bf v; u32x4 q[2]; };
    F fa, fb0, fb1, fb2, fb3;
    fa.q[0]  = *(const u32x4*)&As[cur][arow][abase];
    fa.q[1]  = *(const u32x4*)&As[cur][arow][abase + 8];
    fb0.q[0] = *(const u32x4*)&Bs[cur][ 0 + l16][kb2];
    fb0.q[1] = *(const u32x4*)&Bs[cur][ 0 + l16][kb2 + 8];
    fb1.q[0] = *(const u32x4*)&Bs[cur][16 + l16][kb2];
    fb1.q[1] = *(const u32x4*)&Bs[cur][16 + l16][kb2 + 8];
    fb2.q[0] = *(const u32x4*)&Bs[cur][32 + l16][kb2];
    fb2.q[1] = *(const u32x4*)&Bs[cur][32 + l16][kb2 + 8];
    fb3.q[0] = *(const u32x4*)&Bs[cur][48 + l16][kb2];
    fb3.q[1] = *(const u32x4*)&Bs[cur][48 + l16][kb2 + 8];

    // ---- 4 back-to-back WMMAs (independent accumulators, shared A) ----
    acc[0] = __builtin_amdgcn_wmma_f32_16x16x32_bf16(
        false, fa.v, false, fb0.v, (short)0, acc[0], false, false);
    acc[1] = __builtin_amdgcn_wmma_f32_16x16x32_bf16(
        false, fa.v, false, fb1.v, (short)0, acc[1], false, false);
    acc[2] = __builtin_amdgcn_wmma_f32_16x16x32_bf16(
        false, fa.v, false, fb2.v, (short)0, acc[2], false, false);
    acc[3] = __builtin_amdgcn_wmma_f32_16x16x32_bf16(
        false, fa.v, false, fb3.v, (short)0, acc[3], false, false);

    __syncthreads();
  }

  // ---- epilogue: C/D layout: VGPR r -> row r (lanes 0-15) / r+8 (16-31) ----
  #pragma unroll
  for (int t = 0; t < 4; ++t) {
    const int col = n0 + t * 16 + l16;
    if (col < Nout) {
      const float bv = bias[col];
      #pragma unroll
      for (int r = 0; r < 8; ++r) {
        int row = m0 + wave * 16 + r + half * 8;
        float v = acc[t][r] + bv;
        if (relu) v = fmaxf(v, 0.0f);
        if (outF) outF[(size_t)row * Nout + col] = v;
        if (outB) outB[(size_t)row * Nout + col] = f2bf(v);
      }
    }
  }
}

// ---------------------------------------------------------------------------
// Per-row cosine attention over the gathered relation slice (all f32):
//   rel[n] = bert[scats[n], :, ocats[n], :]   (69 x 512, strided by C*H)
//   cos_r = <vf, rel_r> / max(|vf| |rel_r|, eps); softmax; attended = sum.
// Writes rf = [bf16(vf) | bf16(attended)] (N x 1024) for the next GEMM.
// One block (256 threads = 8 waves) per row.
// ---------------------------------------------------------------------------
__global__ __launch_bounds__(256) void k_attn(
    const float* __restrict__ bert, const float* __restrict__ vf,
    const int* __restrict__ scats, const int* __restrict__ ocats,
    bhalf* __restrict__ rf)
{
  __shared__ float sdot[RREL];
  __shared__ float sss[RREL];
  __shared__ float sattn[RREL];
  __shared__ float svn;

  const int n    = blockIdx.x;
  const int tid  = threadIdx.x;
  const int wave = tid >> 5;
  const int lane = tid & 31;

  const int s = scats[n];
  const int o = ocats[n];
  // element-row index of bert[s, r, o, :] is (s*R*C + r*C + o)
  const size_t baseSO = (size_t)s * RREL * CCAT + (size_t)o;
  const float* vfn = vf + (size_t)n * HDIM;

  // |vf|^2 (wave 0)
  if (wave == 0) {
    float ss = 0.0f;
    #pragma unroll
    for (int i = 0; i < HDIM / 32; ++i) {
      float v = vfn[lane + 32 * i];
      ss += v * v;
    }
    for (int off = 16; off; off >>= 1) ss += __shfl_xor(ss, off, 32);
    if (lane == 0) svn = ss;
  }

  // per-relation dot + squared norm; wave w takes r = w, w+8, ...
  for (int r = wave; r < RREL; r += 8) {
    const float* rp = bert + (baseSO + (size_t)r * CCAT) * HDIM;
    float dot = 0.0f, ss = 0.0f;
    #pragma unroll
    for (int i = 0; i < HDIM / 32; ++i) {
      int d = lane + 32 * i;
      float x = rp[d];
      float v = vfn[d];
      dot += x * v;
      ss  += x * x;
    }
    for (int off = 16; off; off >>= 1) {
      dot += __shfl_xor(dot, off, 32);
      ss  += __shfl_xor(ss, off, 32);
    }
    if (lane == 0) { sdot[r] = dot; sss[r] = ss; }
  }
  __syncthreads();

  // softmax over 69 values (cheap; single thread)
  if (tid == 0) {
    float vn = sqrtf(svn);
    float mx = -1e30f;
    for (int r = 0; r < RREL; ++r) {
      float c = sdot[r] / fmaxf(vn * sqrtf(sss[r]), EPSV);
      sattn[r] = c;
      mx = fmaxf(mx, c);
    }
    float sum = 0.0f;
    for (int r = 0; r < RREL; ++r) {
      float e = __expf(sattn[r] - mx);
      sattn[r] = e;
      sum += e;
    }
    float inv = 1.0f / sum;
    for (int r = 0; r < RREL; ++r) sattn[r] *= inv;
  }
  __syncthreads();

  // attended = sum_r attn_r * rel_r  (coalesced re-read, L2-resident)
  for (int d = tid; d < HDIM; d += 256) {
    const float* rp = bert + baseSO * HDIM + d;
    float acc = 0.0f;
    #pragma unroll 4
    for (int r = 0; r < RREL; ++r)
      acc += sattn[r] * rp[(size_t)r * CCAT * HDIM];
    rf[(size_t)n * 1024 + d]       = f2bf(vfn[d]);
    rf[(size_t)n * 1024 + 512 + d] = f2bf(acc);
  }
}

// ---------------------------------------------------------------------------
// Host-side launch: weight transposes -> GEMM1 -> GEMM2 -> attention ->
// GEMM3 -> GEMM4 (all on `stream`, graph-capture safe).
// ---------------------------------------------------------------------------
extern "C" void kernel_launch(void* const* d_in, const int* in_sizes, int n_in,
                              void* d_out, int out_size, void* d_ws, size_t ws_size,
                              hipStream_t stream) {
  (void)in_sizes; (void)n_in; (void)out_size; (void)ws_size;

  const float* sfeat = (const float*)d_in[0];
  const float* pfeat = (const float*)d_in[1];
  const float* ofeat = (const float*)d_in[2];
  const float* bert  = (const float*)d_in[3];
  const float* w1    = (const float*)d_in[4];
  const float* b1    = (const float*)d_in[5];
  const float* w2    = (const float*)d_in[6];
  const float* b2    = (const float*)d_in[7];
  const float* rw1   = (const float*)d_in[8];
  const float* rb1   = (const float*)d_in[9];
  const float* rw2   = (const float*)d_in[10];
  const float* rb2   = (const float*)d_in[11];
  const int*   scats = (const int*)d_in[12];
  const int*   ocats = (const int*)d_in[13];
  float* out = (float*)d_out;

  // workspace carve-up (256B aligned), ~75 MB total
  char* ws = (char*)d_ws;
  size_t off = 0;
  auto carve = [&](size_t bytes) {
    void* p = ws + off;
    off += (bytes + 255) & ~(size_t)255;
    return p;
  };
  bhalf* vbf  = (bhalf*)carve((size_t)NROWS * 1536 * 2);   // concat bf16
  bhalf* w1t  = (bhalf*)carve((size_t)512  * 1536 * 2);    // [512][1536]
  bhalf* w2t  = (bhalf*)carve((size_t)512  * 512  * 2);    // [512][512]
  bhalf* rw1t = (bhalf*)carve((size_t)512  * 1024 * 2);    // [512][1024]
  bhalf* rw2t = (bhalf*)carve((size_t)128  * 512  * 2);    // [128][512] (69 pad)
  bhalf* hbf  = (bhalf*)carve((size_t)NROWS * 512 * 2);    // relu(v@w1+b1)
  float* vff  = (float*)carve((size_t)NROWS * 512 * 4);    // vf (f32 for attn)
  bhalf* rfb  = (bhalf*)carve((size_t)NROWS * 1024 * 2);   // [vf | attended]
  bhalf* h2b  = (bhalf*)carve((size_t)NROWS * 512 * 2);    // relu(rf@rw1+rb1)

  // weight transpose+convert (one-time, tiny vs. main traffic)
  k_transpose_bf16<<<512, 256, 0, stream>>>(w1,  w1t,  1536, 512, 512);
  k_transpose_bf16<<<256, 256, 0, stream>>>(w2,  w2t,  512,  512, 512);
  k_transpose_bf16<<<256, 256, 0, stream>>>(rw1, rw1t, 1024, 512, 512);
  k_transpose_bf16<<<64,  256, 0, stream>>>(rw2, rw2t, 512,  PDIM, 128);
  k_concat_bf16<<<2048, 256, 0, stream>>>(sfeat, pfeat, ofeat, vbf);

  dim3 gFull(512 / 64, NROWS / 128);            // 8 x 64 blocks
  // GEMM1: h = relu(v @ w1 + b1)   (bf16 out only)
  k_gemm_bf16<<<gFull, 256, 0, stream>>>(vbf, w1t, b1, nullptr, hbf,
                                         NROWS, 1536, 512, 1);
  // GEMM2: vf = relu(h @ w2 + b2)  (f32 out for attention)
  k_gemm_bf16<<<gFull, 256, 0, stream>>>(hbf, w2t, b2, vff, nullptr,
                                         NROWS, 512, 512, 1);
  // attention + rf concat (f32 math, bf16 rf)
  k_attn<<<NROWS, 256, 0, stream>>>(bert, vff, scats, ocats, rfb);
  // GEMM3: h2 = relu(rf @ rw1 + rb1)
  k_gemm_bf16<<<gFull, 256, 0, stream>>>(rfb, rw1t, rb1, nullptr, h2b,
                                         NROWS, 1024, 512, 1);
  // GEMM4: plogits = h2 @ rw2 + rb2  (f32 out, no relu)
  dim3 gLast(128 / 64, NROWS / 128);
  k_gemm_bf16<<<gLast, 256, 0, stream>>>(h2b, rw2t, rb2, out, nullptr,
                                         NROWS, 512, PDIM, 0);
}
